// MultiheadAttention_17480516894981
// MI455X (gfx1250) — matrix operand order repaired and checked by reading, not compiled, not run on hardware
//
#include <hip/hip_runtime.h>
#include <stdint.h>

// ---------------------------------------------------------------------------
// MHA for MI455X (gfx1250, wave32, WMMA). T=2048 B=4 C=D=512 H=8 hd=64.
// Stage 0: one-shot f32->bf16 weight conversion (Wq|Wk|Wv|Wo).
// Stage 1: QKV projections (bf16 WMMA), K stored transposed [b][h][hd][T].
// Stage 2: flash attention; 8 waves/block share LDS-staged K/V tiles
//          (double-buffered), one wave per 16-query tile, key tiles of 32.
// Stage 3: output projection (bf16 WMMA, f32 out).
// ---------------------------------------------------------------------------

typedef __bf16 bf16_t;
typedef bf16_t   v16bf __attribute__((ext_vector_type(16)));
typedef float    v8f   __attribute__((ext_vector_type(8)));
typedef uint32_t v4u   __attribute__((ext_vector_type(4)));

union Frag {
  v16bf    v;
  bf16_t   h[16];
  uint32_t u[8];
  v4u      q[2];
};

__device__ __forceinline__ v8f wmma_bf16(v16bf a, v16bf b, v8f c) {
  // 8 args: (neg_a, A, neg_b, B, c_mod, C, reuse_a, reuse_b)
  return __builtin_amdgcn_wmma_f32_16x16x32_bf16(false, a, false, b,
                                                 (short)0, c, false, false);
}

__device__ __forceinline__ v8f zero8() {
  v8f z = {0.f, 0.f, 0.f, 0.f, 0.f, 0.f, 0.f, 0.f};
  return z;
}

// K index inside a 16x32 bf16 A-fragment for element pair j (0..7), lane half hl.
// Lanes 0-15: VGPR0-3 -> K 0..7, VGPR4-7 -> K 16..23.
// Lanes 16-31: VGPR0-3 -> K 8..15, VGPR4-7 -> K 24..31.
__device__ __forceinline__ int a_k_of(int j, int hl) {
  return (j < 4) ? (hl * 8 + 2 * j) : (16 + hl * 8 + 2 * (j - 4));
}

#define T_DIM  2048
#define B_DIM  4
#define C_DIM  512
#define D_DIM  512
#define H_DIM  8
#define HD_DIM 64
#define M_ROWS (T_DIM * B_DIM)                       // 8192 GEMM rows
#define QKV_ELEMS (B_DIM * H_DIM * T_DIM * HD_DIM)   // 4,194,304
#define W_ELEMS (C_DIM * D_DIM)                      // 262,144 per matrix

// ---------------------------------------------------------------------------
// Stage 0: convert the four weight matrices to bf16, once.
// grid (W_ELEMS/1024, 4), block 256; 4 floats/thread.
// ---------------------------------------------------------------------------
__global__ __launch_bounds__(256) void mha_convert_w_kernel(
    const float* __restrict__ Wq, const float* __restrict__ Wk,
    const float* __restrict__ Wv, const float* __restrict__ Wo,
    bf16_t* __restrict__ wbf)
{
  const int mat = blockIdx.y;                       // 0=q 1=k 2=v 3=o
  const float* src = (mat == 0) ? Wq : (mat == 1) ? Wk : (mat == 2) ? Wv : Wo;
  const int idx = (blockIdx.x * 256 + threadIdx.x) * 4;
  float4 f = *(const float4*)(src + idx);
  bf16_t* dst = wbf + (size_t)mat * W_ELEMS + idx;
  dst[0] = (bf16_t)f.x;
  dst[1] = (bf16_t)f.y;
  dst[2] = (bf16_t)f.z;
  dst[3] = (bf16_t)f.w;
}

// ---------------------------------------------------------------------------
// Stage 1: y = x @ W + b  (x: (T,B,C) f32, W bf16), scatter to bf16 attention
// layouts. blockIdx.z selects Q(0)/K(1)/V(2). Block 256 thr (8 waves); wave w:
// rows [blockIdx.x*128 + 16w, +16), cols [blockIdx.y*64, +64).
// ---------------------------------------------------------------------------
__global__ __launch_bounds__(256) void mha_proj_kernel(
    const float* __restrict__ key_in,  const float* __restrict__ value_in,
    const float* __restrict__ query_in,
    const float* __restrict__ bk, const float* __restrict__ bv,
    const float* __restrict__ bq,
    const bf16_t* __restrict__ wbf,
    bf16_t* __restrict__ qbuf, bf16_t* __restrict__ ktbuf,
    bf16_t* __restrict__ vbuf)
{
  const int which = blockIdx.z;                      // 0=Q 1=K 2=V
  const float* x    = (which == 0) ? query_in : (which == 1) ? key_in : value_in;
  const float* bias = (which == 0) ? bq : (which == 1) ? bk : bv;
  const bf16_t* W   = wbf + (size_t)which * W_ELEMS; // order matches (q,k,v)

  const int wave  = threadIdx.x >> 5;
  const int lane  = threadIdx.x & 31;
  const int hl    = lane >> 4;
  const int l15   = lane & 15;
  const int mBase = blockIdx.x * 128 + wave * 16;
  const int nBase = blockIdx.y * 64;

  const float* xrow = x + (mBase + l15) * C_DIM;     // A row for this lane

  v8f acc[4];
  #pragma unroll
  for (int cf = 0; cf < 4; ++cf) acc[cf] = zero8();

  for (int kb = 0; kb < C_DIM; kb += 32) {
    // A fragment: 16x32 of x, f32 -> bf16 (contiguous pairs -> float2 loads)
    Frag a;
    #pragma unroll
    for (int j = 0; j < 8; ++j) {
      const int k = a_k_of(j, hl);
      float2 f = *(const float2*)(xrow + kb + k);
      a.h[2 * j]     = (bf16_t)f.x;
      a.h[2 * j + 1] = (bf16_t)f.y;
    }
    // B fragments: lane holds N col (nBase + cf*16 + l15), 16 contraction
    // rows in the same K order as A, gathered as bf16.
    const bf16_t* wrow = W + (size_t)kb * D_DIM + nBase + l15;
    #pragma unroll
    for (int cf = 0; cf < 4; ++cf) {
      Frag bf_;
      #pragma unroll
      for (int s = 0; s < 16; ++s) {
        const int k = (s < 8) ? (hl * 8 + s) : (16 + hl * 8 + (s - 8));
        bf_.h[s] = wrow[(size_t)k * D_DIM + cf * 16];
      }
      acc[cf] = wmma_bf16(a.v, bf_.v, acc[cf]);
    }
  }

  // Bias + scatter store (C layout: N = l15, M = r + 8*hl)
  #pragma unroll
  for (int cf = 0; cf < 4; ++cf) {
    const int n   = nBase + cf * 16 + l15;
    const float bb = bias[n];
    const int h   = n >> 6;
    const int dh  = n & 63;
    #pragma unroll
    for (int r = 0; r < 8; ++r) {
      const int m = mBase + hl * 8 + r;
      const int t = m >> 2;                          // row m = t*B + b
      const int b = m & 3;
      const bf16_t o = (bf16_t)(acc[cf][r] + bb);
      const int bh = b * H_DIM + h;
      if (which == 1) {
        ktbuf[(bh * HD_DIM + dh) * T_DIM + t] = o;   // K transposed
      } else if (which == 0) {
        qbuf[(bh * T_DIM + t) * HD_DIM + dh] = o;
      } else {
        vbuf[(bh * T_DIM + t) * HD_DIM + dh] = o;
      }
    }
  }
}

// ---------------------------------------------------------------------------
// Stage 2: flash attention. Block 256 thr = 8 waves of one (b,h); each wave
// owns 16 query rows. Key tiles of 32, cooperatively staged in LDS
// (double-buffered), online softmax. grid = (T/128, B*H).
// ---------------------------------------------------------------------------
__global__ __launch_bounds__(256) void mha_attn_kernel(
    const bf16_t* __restrict__ qbuf, const bf16_t* __restrict__ ktbuf,
    const bf16_t* __restrict__ vbuf, bf16_t* __restrict__ xbuf)
{
  // Kt tile: [hd=64][key 32]; V tile: [key 32][hd 64]; 4 KB each, x2 buffers.
  __shared__ __align__(16) bf16_t kt_s[2][64 * 32];
  __shared__ __align__(16) bf16_t v_s[2][32 * 64];
  __shared__ __align__(16) bf16_t ptile[8][16 * 32]; // per-wave P re-layout

  const int tid  = threadIdx.x;
  const int wave = tid >> 5;
  const int lane = tid & 31;
  const int hl   = lane >> 4;
  const int l15  = lane & 15;

  const int bh    = blockIdx.y;                      // b*H + h
  const int qBase = blockIdx.x * 128 + wave * 16;

  const bf16_t* qp  = qbuf  + ((size_t)bh * T_DIM + qBase) * HD_DIM;
  const bf16_t* ktp = ktbuf + (size_t)bh * HD_DIM * T_DIM;   // [hd][T]
  const bf16_t* vp  = vbuf  + (size_t)bh * T_DIM * HD_DIM;   // [T][hd]

  // Cooperative-staging role: threads 0-127 feed Kt tile, 128-255 feed V tile.
  // Each thread moves 16 bf16 (32 B) per tile: two b128 loads + two b128 LDS
  // stores.
  const bool isK = tid < 128;
  const int  kc  = tid & 127;
  const int  ldsOff = kc * 16;

  // Q A-fragments: hd chunks [0,32) and [32,64); resident for whole loop.
  Frag aq[2];
  #pragma unroll
  for (int f = 0; f < 2; ++f)
    #pragma unroll
    for (int j = 0; j < 8; ++j) {
      const int k = f * 32 + a_k_of(j, hl);
      aq[f].u[j] = *(const uint32_t*)(qp + l15 * HD_DIM + k);
    }

  float mrow[8], lrow[8];
  v8f o[4];
  #pragma unroll
  for (int r = 0; r < 8; ++r) { mrow[r] = -1.0e30f; lrow[r] = 0.f; }
  #pragma unroll
  for (int cf = 0; cf < 4; ++cf) o[cf] = zero8();

  const float scale = 0.125f;                        // 1/sqrt(64)
  const int NT = T_DIM / 32;                         // 64 key tiles

  // Preload tile 0 into buffer 0.
  {
    const bf16_t* g = isK
        ? (ktp + (size_t)(kc >> 1) * T_DIM + ((kc & 1) * 16))
        : (vp + (size_t)(kc >> 2) * HD_DIM + ((kc & 3) * 16));
    v4u r0 = *(const v4u*)g;
    v4u r1 = *(const v4u*)(g + 8);
    bf16_t* d = isK ? &kt_s[0][ldsOff] : &v_s[0][ldsOff];
    *(v4u*)d = r0;
    *(v4u*)(d + 8) = r1;
  }
  __syncthreads();

  for (int kt = 0; kt < NT; ++kt) {
    const int keyBase = kt * 32;
    const int cur = kt & 1;
    const int nxt = cur ^ 1;

    // Issue global loads for the next tile before computing on this one.
    v4u r0 = {0, 0, 0, 0}, r1 = {0, 0, 0, 0};
    if (kt + 1 < NT) {
      const bf16_t* g = isK
          ? (ktp + (size_t)(kc >> 1) * T_DIM + (keyBase + 32) + ((kc & 1) * 16))
          : (vp + (size_t)(keyBase + 32 + (kc >> 2)) * HD_DIM + ((kc & 3) * 16));
      r0 = *(const v4u*)g;
      r1 = *(const v4u*)(g + 8);
    }

    // S = Q Kt for two 16-key groups (B fragments from LDS Kt tile).
    v8f s[2];
    #pragma unroll
    for (int kg = 0; kg < 2; ++kg) {
      v8f accv = zero8();
      #pragma unroll
      for (int f = 0; f < 2; ++f) {
        Frag bk_;
        #pragma unroll
        for (int s = 0; s < 16; ++s) {
          const int k = (s < 8) ? (hl * 8 + s) : (16 + hl * 8 + (s - 8));
          bk_.h[s] = kt_s[cur][(f * 32 + k) * 32 + kg * 16 + l15];
        }
        accv = wmma_bf16(aq[f].v, bk_.v, accv);
      }
      s[kg] = accv;
    }

    // Row max over the 32 keys (reduce inside each 16-lane half).
    float pmax[8];
    #pragma unroll
    for (int r = 0; r < 8; ++r) {
      s[0][r] *= scale;
      s[1][r] *= scale;
      float t = fmaxf(s[0][r], s[1][r]);
      #pragma unroll
      for (int off = 1; off < 16; off <<= 1)
        t = fmaxf(t, __shfl_xor(t, off, 32));
      pmax[r] = t;
    }

    // Online softmax update.
    float alpha[8];
    #pragma unroll
    for (int r = 0; r < 8; ++r) {
      const float mnew = fmaxf(mrow[r], pmax[r]);
      alpha[r] = __expf(mrow[r] - mnew);
      mrow[r]  = mnew;
      const float p0 = __expf(s[0][r] - mnew);
      const float p1 = __expf(s[1][r] - mnew);
      s[0][r] = p0;
      s[1][r] = p1;
      float t = p0 + p1;
      #pragma unroll
      for (int off = 1; off < 16; off <<= 1) t += __shfl_xor(t, off, 32);
      lrow[r] = alpha[r] * lrow[r] + t;
    }

    // C-layout P -> A-layout P via the wave-private LDS tile. Same-wave DS
    // ops are in-order (DScnt); wave_barrier stops compiler reordering.
    bf16_t* pt = ptile[wave];
    #pragma unroll
    for (int kg = 0; kg < 2; ++kg)
      #pragma unroll
      for (int r = 0; r < 8; ++r)
        pt[(r + 8 * hl) * 32 + kg * 16 + l15] = (bf16_t)s[kg][r];
    __builtin_amdgcn_wave_barrier();
    Frag pA;
    #pragma unroll
    for (int j = 0; j < 8; ++j) {
      const int k = a_k_of(j, hl);
      pA.u[j] = *(const uint32_t*)(pt + l15 * 32 + k);
    }
    __builtin_amdgcn_wave_barrier();

    // O = diag(alpha) * O + P @ V (B fragments from LDS V tile).
    #pragma unroll
    for (int cf = 0; cf < 4; ++cf) {
      v8f c = o[cf];
      #pragma unroll
      for (int r = 0; r < 8; ++r) c[r] *= alpha[r];
      Frag bv_;
      #pragma unroll
      for (int s = 0; s < 16; ++s) {
        const int k = (s < 8) ? (hl * 8 + s) : (16 + hl * 8 + (s - 8));
        bv_.h[s] = v_s[cur][k * 64 + cf * 16 + l15];
      }
      o[cf] = wmma_bf16(pA.v, bv_.v, c);
    }

    // Publish the next tile and synchronize the block.
    if (kt + 1 < NT) {
      bf16_t* d = isK ? &kt_s[nxt][ldsOff] : &v_s[nxt][ldsOff];
      *(v4u*)d = r0;
      *(v4u*)(d + 8) = r1;
    }
    __syncthreads();
  }

  // Normalize and store to X in (T,B,D) bf16; d = h*64 + hd-col.
  const int b = bh >> 3;
  const int h = bh & 7;
  #pragma unroll
  for (int cf = 0; cf < 4; ++cf) {
    const int n = cf * 16 + l15;
    #pragma unroll
    for (int r = 0; r < 8; ++r) {
      const int t   = qBase + r + 8 * hl;
      const float v = o[cf][r] * (1.0f / lrow[r]);
      xbuf[((size_t)t * B_DIM + b) * D_DIM + h * HD_DIM + n] = (bf16_t)v;
    }
  }
}

// ---------------------------------------------------------------------------
// Stage 3: out = X @ Wo + bo, X bf16 (T,B,D), Wo bf16, out f32 (T,B,D).
// ---------------------------------------------------------------------------
__global__ __launch_bounds__(256) void mha_outproj_kernel(
    const bf16_t* __restrict__ X, const bf16_t* __restrict__ wbf,
    const float* __restrict__ bo, float* __restrict__ out)
{
  const int wave  = threadIdx.x >> 5;
  const int lane  = threadIdx.x & 31;
  const int hl    = lane >> 4;
  const int l15   = lane & 15;
  const int mBase = blockIdx.x * 128 + wave * 16;
  const int nBase = blockIdx.y * 64;

  const bf16_t* Wo = wbf + (size_t)3 * W_ELEMS;
  const bf16_t* xrow = X + (size_t)(mBase + l15) * D_DIM;

  v8f acc[4];
  #pragma unroll
  for (int cf = 0; cf < 4; ++cf) acc[cf] = zero8();

  for (int kb = 0; kb < D_DIM; kb += 32) {
    Frag a;
    #pragma unroll
    for (int j = 0; j < 8; ++j) {
      const int k = a_k_of(j, hl);
      a.u[j] = *(const uint32_t*)(xrow + kb + k);
    }
    const bf16_t* wrow = Wo + (size_t)kb * D_DIM + nBase + l15;
    #pragma unroll
    for (int cf = 0; cf < 4; ++cf) {
      Frag bf_;
      #pragma unroll
      for (int s = 0; s < 16; ++s) {
        const int k = (s < 8) ? (hl * 8 + s) : (16 + hl * 8 + (s - 8));
        bf_.h[s] = wrow[(size_t)k * D_DIM + cf * 16];
      }
      acc[cf] = wmma_bf16(a.v, bf_.v, acc[cf]);
    }
  }

  #pragma unroll
  for (int cf = 0; cf < 4; ++cf) {
    const int n   = nBase + cf * 16 + l15;
    const float bb = bo[n];
    #pragma unroll
    for (int r = 0; r < 8; ++r) {
      const int m = mBase + hl * 8 + r;
      out[(size_t)m * D_DIM + n] = acc[cf][r] + bb;
    }
  }
}

// ---------------------------------------------------------------------------
extern "C" void kernel_launch(void* const* d_in, const int* in_sizes, int n_in,
                              void* d_out, int out_size, void* d_ws,
                              size_t ws_size, hipStream_t stream) {
  const float* key_in   = (const float*)d_in[0];
  const float* value_in = (const float*)d_in[1];
  const float* query_in = (const float*)d_in[2];
  const float* Wk = (const float*)d_in[3];
  const float* bk = (const float*)d_in[4];
  const float* Wv = (const float*)d_in[5];
  const float* bv = (const float*)d_in[6];
  const float* Wq = (const float*)d_in[7];
  const float* bq = (const float*)d_in[8];
  const float* Wo = (const float*)d_in[9];
  const float* bo = (const float*)d_in[10];

  // Workspace (bf16): Q | Kt | V | X (8 MB each) | W bf16 (2 MB) = 34 MB.
  bf16_t* qbuf  = (bf16_t*)d_ws;
  bf16_t* ktbuf = qbuf + QKV_ELEMS;
  bf16_t* vbuf  = ktbuf + QKV_ELEMS;
  bf16_t* xbuf  = vbuf + QKV_ELEMS;
  bf16_t* wbf   = xbuf + QKV_ELEMS;

  dim3 blk(256, 1, 1);

  dim3 gCvt(W_ELEMS / 1024, 4, 1);                  // 256 x 4
  mha_convert_w_kernel<<<gCvt, blk, 0, stream>>>(Wq, Wk, Wv, Wo, wbf);

  dim3 gProj(M_ROWS / 128, D_DIM / 64, 3);          // 64 x 8 x 3
  mha_proj_kernel<<<gProj, blk, 0, stream>>>(
      key_in, value_in, query_in, bk, bv, bq, wbf, qbuf, ktbuf, vbuf);

  dim3 gAttn(T_DIM / 128, B_DIM * H_DIM, 1);        // 16 x 32
  mha_attn_kernel<<<gAttn, blk, 0, stream>>>(qbuf, ktbuf, vbuf, xbuf);

  dim3 gOut(M_ROWS / 128, D_DIM / 64, 1);           // 64 x 8
  mha_outproj_kernel<<<gOut, blk, 0, stream>>>(xbuf, wbf, bo, (float*)d_out);
}
